// MixedAttention_6932077215855
// MI455X (gfx1250) — matrix operand order repaired and checked by reading, not compiled, not run on hardware
//
#include <hip/hip_runtime.h>

typedef __bf16 bf16_t;
typedef bf16_t v16bf __attribute__((ext_vector_type(16)));
typedef float  v8f   __attribute__((ext_vector_type(8)));
typedef unsigned short u16;
typedef unsigned int   u32;

// ---------- bf16 helpers (manual RNE conversion; storage as ushort) ----------
__device__ __forceinline__ u16 f2b(float f) {
    u32 u = __float_as_uint(f);
    u32 r = u + 0x7FFFu + ((u >> 16) & 1u);
    return (u16)(r >> 16);
}
__device__ __forceinline__ float b2f(u16 h) {
    return __uint_as_float(((u32)h) << 16);
}

union BFrag { v16bf v; u16 u[16]; uint4 q[2]; };
union V4U8  { uint4 q; u16 u[8]; };

__device__ __forceinline__ v8f v8f_zero() {
    v8f v;
#pragma unroll
    for (int i = 0; i < 8; ++i) v[i] = 0.f;
    return v;
}

// B fragment (32x16, K x N): lane holds col = lane&15; elements K = 16*hl + i.
// p must be 16B aligned; 16 contiguous bf16 -> two b128 loads.
__device__ __forceinline__ v16bf load_frag_b16(const u16* p) {
    BFrag f;
    f.q[0] = *reinterpret_cast<const uint4*>(p);
    f.q[1] = *reinterpret_cast<const uint4*>(p + 8);
    return f.v;
}

// A fragment (16x32, M x K): lane r = lane&15 holds row r; hl = lane>>4.
// Element i -> K = 16*hl + i.
__device__ __forceinline__ v16bf load_frag_a(const u16* row, int hl) {
    BFrag f;
    f.q[0] = *reinterpret_cast<const uint4*>(row + 16 * hl);
    f.q[1] = *reinterpret_cast<const uint4*>(row + 16 * hl + 8);
    return f.v;
}

// A fragment with L2 row-normalization (partner lane lane^16 holds other 16 dims)
__device__ __forceinline__ v16bf load_a_norm(const u16* row, int hl) {
    V4U8 t0, t1;
    t0.q = *reinterpret_cast<const uint4*>(row + 16 * hl);
    t1.q = *reinterpret_cast<const uint4*>(row + 16 * hl + 8);
    float f[16];
    float ss = 0.f;
#pragma unroll
    for (int i = 0; i < 8; ++i) { f[i] = b2f(t0.u[i]); ss += f[i] * f[i]; }
#pragma unroll
    for (int i = 0; i < 8; ++i) { f[8 + i] = b2f(t1.u[i]); ss += f[8 + i] * f[8 + i]; }
    ss += __shfl_xor(ss, 16, 32);
    float sc = 1.0f / fmaxf(sqrtf(ss), 1e-12f);
    BFrag r;
#pragma unroll
    for (int i = 0; i < 16; ++i) r.u[i] = f2b(f[i] * sc);
    return r.v;
}

__device__ __forceinline__ v8f wmma_bf16(v16bf a, v16bf b, v8f c) {
    return __builtin_amdgcn_wmma_f32_16x16x32_bf16(false, a, false, b, (short)0, c, false, false);
}

// Load 8 consecutive f32 bias values (32B aligned) as two b128 loads.
__device__ __forceinline__ void load_bias8(const float* p, float* bb) {
    float4 b0 = reinterpret_cast<const float4*>(p)[0];
    float4 b1 = reinterpret_cast<const float4*>(p)[1];
    bb[0] = b0.x; bb[1] = b0.y; bb[2] = b0.z; bb[3] = b0.w;
    bb[4] = b1.x; bb[5] = b1.y; bb[6] = b1.z; bb[7] = b1.w;
}

// Load a 32-dim bf16 row (64B, 16B aligned), normalize, store 64B to LDS row.
__device__ __forceinline__ void norm_row_to_lds(const u16* src, u16* dst) {
    V4U8 t[4];
    const uint4* sq = reinterpret_cast<const uint4*>(src);
#pragma unroll
    for (int i = 0; i < 4; ++i) t[i].q = sq[i];
    float f[32], ss = 0.f;
#pragma unroll
    for (int i = 0; i < 4; ++i)
#pragma unroll
        for (int j = 0; j < 8; ++j) { f[i * 8 + j] = b2f(t[i].u[j]); ss += f[i * 8 + j] * f[i * 8 + j]; }
    float sc = 1.f / fmaxf(sqrtf(ss), 1e-12f);
    uint4* dq = reinterpret_cast<uint4*>(dst);
#pragma unroll
    for (int i = 0; i < 4; ++i) {
        V4U8 o;
#pragma unroll
        for (int j = 0; j < 8; ++j) o.u[j] = f2b(f[i * 8 + j] * sc);
        dq[i] = o.q;
    }
}

// =============================== small kernels ===============================
__global__ void cvt_bf16_k(const float* __restrict__ in, u16* __restrict__ out, long n) {
    long i = ((long)blockIdx.x * blockDim.x + threadIdx.x) * 8;
    if (i >= n) return;
    if (i + 8 <= n) {
        float4 a = *reinterpret_cast<const float4*>(in + i);
        float4 b = *reinterpret_cast<const float4*>(in + i + 4);
        V4U8 o;
        o.u[0] = f2b(a.x); o.u[1] = f2b(a.y); o.u[2] = f2b(a.z); o.u[3] = f2b(a.w);
        o.u[4] = f2b(b.x); o.u[5] = f2b(b.y); o.u[6] = f2b(b.z); o.u[7] = f2b(b.w);
        *reinterpret_cast<uint4*>(out + i) = o.q;
    } else {
        for (long k = i; k < n; ++k) out[k] = f2b(in[k]);
    }
}

// AvgPool2d(4) over (256,256,256) -> pooled bf16 (4096,256)
__global__ void pool_k(const float* __restrict__ x, u16* __restrict__ pooled) {
    int p = blockIdx.x;           // 0..4095  (py*64+px)
    int c = threadIdx.x;          // 0..255
    int py = p >> 6, px = p & 63;
    float a = 0.f;
#pragma unroll
    for (int i = 0; i < 4; ++i)
#pragma unroll
        for (int j = 0; j < 4; ++j)
            a += x[(((size_t)(py * 4 + i)) * 256 + (px * 4 + j)) * 256 + c];
    pooled[(size_t)p * 256 + c] = f2b(a * 0.0625f);
}

// bt = relu(table(rows,2) @ w1(2,512) + b1) @ w2(512,4)
__global__ void cpb_mlp_k(const float* __restrict__ table, const float* __restrict__ w1,
                          const float* __restrict__ b1, const float* __restrict__ w2,
                          float* __restrict__ bt, int rows) {
    int r = blockIdx.x * blockDim.x + threadIdx.x;
    if (r >= rows) return;
    float t0 = table[2 * r], t1 = table[2 * r + 1];
    float a0 = 0.f, a1 = 0.f, a2 = 0.f, a3 = 0.f;
    for (int j = 0; j < 512; ++j) {
        float hg = fmaxf(t0 * w1[j] + t1 * w1[512 + j] + b1[j], 0.f);
        a0 += hg * w2[j * 4 + 0];
        a1 += hg * w2[j * 4 + 1];
        a2 += hg * w2[j * 4 + 2];
        a3 += hg * w2[j * 4 + 3];
    }
    bt[r * 4 + 0] = a0; bt[r * 4 + 1] = a1;
    bt[r * 4 + 2] = a2; bt[r * 4 + 3] = a3;
}

// TRANSPOSED bias expansion: out[h][col][row] = 16*sigmoid(bt[index[row*N2+col]][h])
// (attn bias matrix is N1 rows x N2 cols; we store column-major per head so the
//  attention kernels can read 8 consecutive rows per lane with b128 loads)
__global__ void bias_expand_k(const float* __restrict__ bt, const int* __restrict__ index,
                              float* __restrict__ out, int N1, int N2) {
    int n1n2 = N1 * N2;
    int i = blockIdx.x * blockDim.x + threadIdx.x;
    if (i >= 4 * n1n2) return;
    int h = i / n1n2, r = i - h * n1n2;
    int col = r / N1, row = r - col * N1;
    float v = bt[index[row * N2 + col] * 4 + h];
    out[i] = 16.f / (1.f + __expf(-v));
}

// =============================== WMMA GEMM ===============================
// C(MxN,f32) = A(MxK,bf16) @ B(KxN,bf16) + bias(N) ; out bf16 or f32.
// Block tile 128x64, 8 waves (256 thr), each wave 32x32 (2x2 wmma tiles).
template <bool OUT_F32>
__global__ __launch_bounds__(256) void gemm_bf16_k(
    const u16* __restrict__ A, const u16* __restrict__ B,
    const float* __restrict__ bias, void* __restrict__ out,
    int M, int N, int K) {
    __shared__ alignas(16) u16 As[128 * 32];   // row-major MxK tile
    __shared__ alignas(16) u16 BsT[64 * 32];   // transposed: [n][k]
    int tid = threadIdx.x;
    int lane = tid & 31, wid = tid >> 5;
    int hl = lane >> 4, ln = lane & 15;
    int m0 = blockIdx.x * 128, n0 = blockIdx.y * 64;
    int wm = wid >> 1, wn = wid & 1;

    v8f acc[2][2];
#pragma unroll
    for (int i = 0; i < 2; ++i)
#pragma unroll
        for (int j = 0; j < 2; ++j) acc[i][j] = v8f_zero();

    for (int k0 = 0; k0 < K; k0 += 32) {
        __syncthreads();
        {   // A tile: 128 rows x 32 k ; thread -> 16 contiguous bf16 (2 x b128)
            int r = tid >> 1, cs = (tid & 1) * 16;
            const uint4* src = reinterpret_cast<const uint4*>(A + (size_t)(m0 + r) * K + k0 + cs);
            uint4* dst = reinterpret_cast<uint4*>(As + r * 32 + cs);
            dst[0] = src[0];
            dst[1] = src[1];
            if (k0 + 32 < K) __builtin_prefetch(src + 4, 0, 1);
        }
        {   // B tile: 32 k x 64 n, transpose into BsT (b128 load, scatter ds stores)
            int r = tid >> 3, cs = (tid & 7) * 8;
            V4U8 t;
            t.q = *reinterpret_cast<const uint4*>(B + (size_t)(k0 + r) * N + n0 + cs);
#pragma unroll
            for (int i = 0; i < 8; ++i) BsT[(cs + i) * 32 + r] = t.u[i];
        }
        __syncthreads();
        v16bf a[2], b[2];
#pragma unroll
        for (int i = 0; i < 2; ++i)
            a[i] = load_frag_a(&As[(wm * 32 + i * 16 + ln) * 32], hl);
#pragma unroll
        for (int j = 0; j < 2; ++j)
            b[j] = load_frag_b16(&BsT[(wn * 32 + j * 16 + ln) * 32 + 16 * hl]);
#pragma unroll
        for (int i = 0; i < 2; ++i)
#pragma unroll
            for (int j = 0; j < 2; ++j) acc[i][j] = wmma_bf16(a[i], b[j], acc[i][j]);
    }
#pragma unroll
    for (int i = 0; i < 2; ++i)
#pragma unroll
        for (int j = 0; j < 2; ++j) {
            int col = n0 + wn * 32 + j * 16 + ln;
            float bv = bias ? bias[col] : 0.f;
#pragma unroll
            for (int e = 0; e < 8; ++e) {
                int row = m0 + wm * 32 + i * 16 + e + 8 * hl;
                float v = acc[i][j][e] + bv;
                if (OUT_F32) ((float*)out)[(size_t)row * N + col] = v;
                else         ((u16*)out)[(size_t)row * N + col] = f2b(v);
            }
        }
}

// =============================== window attention ===============================
// grid = 1024 : (window w = b>>2, head h = b&3). 4 waves; wave handles 4 blocks of
// 16 q-rows. S = qn @ kn^T (16x256), P = exp(S*scale + bias), XW = P@V / rowsum.
// biasT layout: [h][key m][query n]
__global__ __launch_bounds__(128) void win_attn_k(
    const u16* __restrict__ qkv, const float* __restrict__ biasT,
    const float* __restrict__ ls, u16* __restrict__ y) {
    __shared__ alignas(16) u16 kn[256 * 32];      // normalized k, row-major
    __shared__ alignas(16) u16 vT[32 * 256];      // v transposed [d][n]
    __shared__ alignas(16) u16 P[4][16 * 128];    // per-wave exp(S) half-row-block
    __shared__ float rs[4][16];
    int b = blockIdx.x;
    int w = b >> 2, h = b & 3;
    int wy = w >> 4, wx = w & 15;
    int tid = threadIdx.x, lane = tid & 31, wid = tid >> 5;
    int hl = lane >> 4, ln = lane & 15;
    float scale = __expf(fminf(ls[h], 4.6051702f));

    for (int row = tid; row < 256; row += 128) {
        int yy = wy * 16 + (row >> 4), xx = wx * 16 + (row & 15);
        size_t L = (size_t)yy * 256 + xx;
        norm_row_to_lds(qkv + L * 768 + 128 + h * 32, &kn[row * 32]);
        const uint4* vq = reinterpret_cast<const uint4*>(qkv + L * 768 + 256 + h * 32);
#pragma unroll
        for (int i = 0; i < 4; ++i) {
            V4U8 t; t.q = vq[i];
#pragma unroll
            for (int j = 0; j < 8; ++j) vT[(i * 8 + j) * 256 + row] = t.u[j];
        }
    }
    __syncthreads();

    const float* bh = biasT + (size_t)h * 256 * 256;
    for (int rbl = 0; rbl < 4; ++rbl) {
        int rb = rbl * 4 + wid, row0 = rb * 16;
        int nq = row0 + ln;
        int yq = wy * 16 + (nq >> 4), xq = wx * 16 + (nq & 15);
        v16bf aq = load_a_norm(qkv + ((size_t)yq * 256 + xq) * 768 + h * 32, hl);
        v8f acc0 = v8f_zero(), acc1 = v8f_zero();
        float rsum = 0.f;
        for (int half = 0; half < 2; ++half) {
#pragma unroll
            for (int ct = 0; ct < 8; ++ct) {
                int col0 = half * 128 + ct * 16;
                v16bf bk = load_frag_b16(&kn[(col0 + ln) * 32 + 16 * hl]);
                v8f s = wmma_bf16(aq, bk, v8f_zero());
                float bb[8];
                load_bias8(bh + (size_t)(col0 + ln) * 256 + row0 + 8 * hl, bb);
#pragma unroll
                for (int e = 0; e < 8; ++e) {
                    int rloc = e + 8 * hl;
                    float p = __expf(s[e] * scale + bb[e]);
                    P[wid][rloc * 128 + ct * 16 + ln] = f2b(p);
                }
            }
            if (lane < 16) {
                float a = 0.f;
                for (int c = 0; c < 128; c += 8) {
                    V4U8 t; t.q = *reinterpret_cast<const uint4*>(&P[wid][lane * 128 + c]);
#pragma unroll
                    for (int j = 0; j < 8; ++j) a += b2f(t.u[j]);
                }
                rsum += a;
            }
#pragma unroll
            for (int ks = 0; ks < 4; ++ks) {
                v16bf ap = load_frag_a(&P[wid][ln * 128 + ks * 32], hl);
                int kg = half * 128 + ks * 32;
                v16bf bv0 = load_frag_b16(&vT[(ln) * 256 + kg + 16 * hl]);
                v16bf bv1 = load_frag_b16(&vT[(16 + ln) * 256 + kg + 16 * hl]);
                acc0 = wmma_bf16(ap, bv0, acc0);
                acc1 = wmma_bf16(ap, bv1, acc1);
            }
        }
        if (lane < 16) rs[wid][lane] = rsum;
#pragma unroll
        for (int e = 0; e < 8; ++e) {
            int rloc = e + 8 * hl;
            float inv = 1.f / rs[wid][rloc];
            int n = row0 + rloc;
            int yo = wy * 16 + (n >> 4), xo = wx * 16 + (n & 15);
            size_t Lo = (size_t)yo * 256 + xo;
            y[Lo * 256 + h * 32 + ln]      = f2b(acc0[e] * inv);
            y[Lo * 256 + h * 32 + 16 + ln] = f2b(acc1[e] * inv);
        }
    }
}

// =============================== stripe pass 1 ===============================
// grid = 256 : (stripe s = b>>2, head h = b&3). a1 = anc_n @ kn^T (64x1024),
// xm = softmax(a1) @ v (64x32). K/V streamed in 8 tiles of 128 rows.
// biasT layout: [h][token m][anchor n2]  (stride 64)
__global__ __launch_bounds__(128) void stripe_p1_k(
    const u16* __restrict__ qkv, const u16* __restrict__ anchor,
    const float* __restrict__ biasT, const float* __restrict__ ls,
    u16* __restrict__ xm) {
    __shared__ alignas(16) u16 kn[128 * 32];
    __shared__ alignas(16) u16 vT[32 * 128];
    __shared__ alignas(16) u16 P[4][16 * 128];
    __shared__ float rs[4][16];
    int b = blockIdx.x;
    int s = b >> 2, h = b & 3;
    int sy = s >> 3, sx = s & 7;
    int tid = threadIdx.x, lane = tid & 31, wid = tid >> 5;
    int hl = lane >> 4, ln = lane & 15;
    float scale = __expf(fminf(ls[h], 4.6051702f));

    int n2 = wid * 16 + ln;
    int ay = sy * 8 + (n2 >> 3), ax = sx * 8 + (n2 & 7);
    v16bf aa = load_a_norm(anchor + ((size_t)ay * 64 + ax) * 128 + h * 32, hl);

    v8f acc0 = v8f_zero(), acc1 = v8f_zero();
    float rsum = 0.f;
    const float* bh = biasT + (size_t)h * 64 * 1024;

    for (int it = 0; it < 8; ++it) {
        __syncthreads();
        {
            int rr = it * 128 + tid;
            int yy = sy * 32 + (rr >> 5), xx = sx * 32 + (rr & 31);
            size_t L = (size_t)yy * 256 + xx;
            norm_row_to_lds(qkv + L * 768 + 512 + h * 32, &kn[tid * 32]);
            const uint4* vq = reinterpret_cast<const uint4*>(qkv + L * 768 + 640 + h * 32);
#pragma unroll
            for (int i = 0; i < 4; ++i) {
                V4U8 t; t.q = vq[i];
#pragma unroll
                for (int j = 0; j < 8; ++j) vT[(i * 8 + j) * 128 + tid] = t.u[j];
            }
        }
        __syncthreads();
#pragma unroll
        for (int ct = 0; ct < 8; ++ct) {
            v16bf bk = load_frag_b16(&kn[(ct * 16 + ln) * 32 + 16 * hl]);
            v8f sA = wmma_bf16(aa, bk, v8f_zero());
            int m = it * 128 + ct * 16 + ln;
            float bb[8];
            load_bias8(bh + (size_t)m * 64 + wid * 16 + 8 * hl, bb);
#pragma unroll
            for (int e = 0; e < 8; ++e) {
                int rloc = e + 8 * hl;
                float p = __expf(sA[e] * scale + bb[e]);
                P[wid][rloc * 128 + ct * 16 + ln] = f2b(p);
            }
        }
        if (lane < 16) {
            float a = 0.f;
            for (int c = 0; c < 128; c += 8) {
                V4U8 t; t.q = *reinterpret_cast<const uint4*>(&P[wid][lane * 128 + c]);
#pragma unroll
                for (int j = 0; j < 8; ++j) a += b2f(t.u[j]);
            }
            rsum += a;
        }
#pragma unroll
        for (int ks = 0; ks < 4; ++ks) {
            v16bf ap = load_frag_a(&P[wid][ln * 128 + ks * 32], hl);
            v16bf bv0 = load_frag_b16(&vT[(ln) * 128 + ks * 32 + 16 * hl]);
            v16bf bv1 = load_frag_b16(&vT[(16 + ln) * 128 + ks * 32 + 16 * hl]);
            acc0 = wmma_bf16(ap, bv0, acc0);
            acc1 = wmma_bf16(ap, bv1, acc1);
        }
    }
    if (lane < 16) rs[wid][lane] = rsum;
#pragma unroll
    for (int e = 0; e < 8; ++e) {
        int rloc = e + 8 * hl;
        float inv = 1.f / rs[wid][rloc];
        int n2o = wid * 16 + rloc;
        size_t base = (((size_t)s * 4 + h) * 64 + n2o) * 32;
        xm[base + ln]      = f2b(acc0[e] * inv);
        xm[base + 16 + ln] = f2b(acc1[e] * inv);
    }
}

// =============================== stripe pass 2 ===============================
// grid = 256 : a2 = qn @ anc_n^T (1024x64), xs = softmax(a2) @ xm (1024x32).
// biasT layout: [h][anchor m][token n1]  (stride 1024)
__global__ __launch_bounds__(128) void stripe_p2_k(
    const u16* __restrict__ qkv, const u16* __restrict__ anchor,
    const u16* __restrict__ xm, const float* __restrict__ biasT,
    const float* __restrict__ ls, u16* __restrict__ y) {
    __shared__ alignas(16) u16 ancn[64 * 32];
    __shared__ alignas(16) u16 xmT[32 * 64];
    __shared__ alignas(16) u16 P[4][16 * 64];
    __shared__ float rs[4][16];
    int b = blockIdx.x;
    int s = b >> 2, h = b & 3;
    int sy = s >> 3, sx = s & 7;
    int tid = threadIdx.x, lane = tid & 31, wid = tid >> 5;
    int hl = lane >> 4, ln = lane & 15;
    float scale = __expf(fminf(ls[h], 4.6051702f));

    if (tid < 64) {
        int ay = sy * 8 + (tid >> 3), ax = sx * 8 + (tid & 7);
        norm_row_to_lds(anchor + ((size_t)ay * 64 + ax) * 128 + h * 32, &ancn[tid * 32]);
    }
    {
        size_t base = ((size_t)s * 4 + h) * 64 * 32;
        for (int c = tid; c < 256; c += 128) {   // 256 chunks of 8
            int n2i = c >> 2, d0 = (c & 3) * 8;
            V4U8 t; t.q = *reinterpret_cast<const uint4*>(&xm[base + (size_t)c * 8]);
#pragma unroll
            for (int j = 0; j < 8; ++j) xmT[(d0 + j) * 64 + n2i] = t.u[j];
        }
    }
    __syncthreads();

    const float* bh = biasT + (size_t)h * 1024 * 64;
    for (int rbl = 0; rbl < 16; ++rbl) {
        int rb = rbl * 4 + wid, row0 = rb * 16;
        int nq = row0 + ln;
        int yq = sy * 32 + (nq >> 5), xq = sx * 32 + (nq & 31);
        v16bf aq = load_a_norm(qkv + ((size_t)yq * 256 + xq) * 768 + 384 + h * 32, hl);
        v8f acc0 = v8f_zero(), acc1 = v8f_zero();
#pragma unroll
        for (int ct = 0; ct < 4; ++ct) {
            v16bf bk = load_frag_b16(&ancn[(ct * 16 + ln) * 32 + 16 * hl]);
            v8f sA = wmma_bf16(aq, bk, v8f_zero());
            float bb[8];
            load_bias8(bh + (size_t)(ct * 16 + ln) * 1024 + row0 + 8 * hl, bb);
#pragma unroll
            for (int e = 0; e < 8; ++e) {
                int rloc = e + 8 * hl;
                float p = __expf(sA[e] * scale + bb[e]);
                P[wid][rloc * 64 + ct * 16 + ln] = f2b(p);
            }
        }
        if (lane < 16) {
            float a = 0.f;
            for (int c = 0; c < 64; c += 8) {
                V4U8 t; t.q = *reinterpret_cast<const uint4*>(&P[wid][lane * 64 + c]);
#pragma unroll
                for (int j = 0; j < 8; ++j) a += b2f(t.u[j]);
            }
            rs[wid][lane] = a;
        }
#pragma unroll
        for (int ks = 0; ks < 2; ++ks) {
            v16bf ap2 = load_frag_a(&P[wid][ln * 64 + ks * 32], hl);
            v16bf bv0 = load_frag_b16(&xmT[(ln) * 64 + ks * 32 + 16 * hl]);
            v16bf bv1 = load_frag_b16(&xmT[(16 + ln) * 64 + ks * 32 + 16 * hl]);
            acc0 = wmma_bf16(ap2, bv0, acc0);
            acc1 = wmma_bf16(ap2, bv1, acc1);
        }
#pragma unroll
        for (int e = 0; e < 8; ++e) {
            int rloc = e + 8 * hl;
            float inv = 1.f / rs[wid][rloc];
            int n = row0 + rloc;
            int yo = sy * 32 + (n >> 5), xo = sx * 32 + (n & 31);
            size_t Lo = (size_t)yo * 256 + xo;
            y[Lo * 256 + 128 + h * 32 + ln]      = f2b(acc0[e] * inv);
            y[Lo * 256 + 128 + h * 32 + 16 + ln] = f2b(acc1[e] * inv);
        }
    }
}

// =============================== launch ===============================
extern "C" void kernel_launch(void* const* d_in, const int* in_sizes, int n_in,
                              void* d_out, int out_size, void* d_ws, size_t ws_size,
                              hipStream_t stream) {
    (void)in_sizes; (void)n_in; (void)out_size; (void)ws_size;
    const float* x        = (const float*)d_in[0];
    const float* qkv_w    = (const float*)d_in[1];
    const float* qkv_b    = (const float*)d_in[2];
    const float* anchor_w = (const float*)d_in[3];
    const float* anchor_b = (const float*)d_in[4];
    const float* ls_w     = (const float*)d_in[5];
    const float* cw1_w    = (const float*)d_in[6];
    const float* cb1_w    = (const float*)d_in[7];
    const float* cw2_w    = (const float*)d_in[8];
    const float* ls_s1    = (const float*)d_in[9];
    const float* cw1_s1   = (const float*)d_in[10];
    const float* cb1_s1   = (const float*)d_in[11];
    const float* cw2_s1   = (const float*)d_in[12];
    const float* ls_s2    = (const float*)d_in[13];
    const float* cw1_s2   = (const float*)d_in[14];
    const float* cb1_s2   = (const float*)d_in[15];
    const float* cw2_s2   = (const float*)d_in[16];
    const float* proj_w   = (const float*)d_in[17];
    const float* proj_b   = (const float*)d_in[18];
    const float* table_w  = (const float*)d_in[19];
    const float* table_s  = (const float*)d_in[20];
    const int* index_w    = (const int*)d_in[21];
    const int* index_a2w  = (const int*)d_in[22];
    const int* index_w2a  = (const int*)d_in[23];

    char* ws = (char*)d_ws;
    size_t off = 0;
    auto alloc = [&](size_t bytes) -> char* {
        char* p = ws + off;
        off += (bytes + 255) & ~(size_t)255;
        return p;
    };
    u16* x_b     = (u16*)alloc(65536ull * 256 * 2);
    u16* wqkv_b  = (u16*)alloc(256ull * 768 * 2);
    u16* wproj_b = (u16*)alloc(256ull * 256 * 2);
    u16* wanc_b  = (u16*)alloc(256ull * 128 * 2);
    u16* qkv     = (u16*)alloc(65536ull * 768 * 2);
    u16* pooled  = (u16*)alloc(4096ull * 256 * 2);
    u16* anchor  = (u16*)alloc(4096ull * 128 * 2);
    u16* yb      = (u16*)alloc(65536ull * 256 * 2);
    u16* xm      = (u16*)alloc(64ull * 4 * 64 * 32 * 2);
    float* bt_w     = (float*)alloc(961ull * 4 * 4);
    float* bt_s1    = (float*)alloc(1521ull * 4 * 4);
    float* bt_s2    = (float*)alloc(1521ull * 4 * 4);
    float* bias_w   = (float*)alloc(4ull * 256 * 256 * 4);
    float* bias_a2w = (float*)alloc(4ull * 64 * 1024 * 4);
    float* bias_w2a = (float*)alloc(4ull * 1024 * 64 * 4);

    // fp32 -> bf16 conversions (8 elems/thread, b128 traffic)
    cvt_bf16_k<<<(65536 * 256 / 8 + 255) / 256, 256, 0, stream>>>(x, x_b, 65536L * 256);
    cvt_bf16_k<<<(256 * 768 / 8 + 255) / 256, 256, 0, stream>>>(qkv_w, wqkv_b, 256L * 768);
    cvt_bf16_k<<<(256 * 256 / 8 + 255) / 256, 256, 0, stream>>>(proj_w, wproj_b, 256L * 256);
    cvt_bf16_k<<<(256 * 128 / 8 + 255) / 256, 256, 0, stream>>>(anchor_w, wanc_b, 256L * 128);

    // qkv = x @ qkv_w + qkv_b  (65536x768)
    gemm_bf16_k<false><<<dim3(512, 12), 256, 0, stream>>>(x_b, wqkv_b, qkv_b, qkv, 65536, 768, 256);

    // anchor = AvgPool4(x) @ anchor_w + anchor_b  (4096x128)
    pool_k<<<4096, 256, 0, stream>>>(x, pooled);
    gemm_bf16_k<false><<<dim3(32, 2), 256, 0, stream>>>(pooled, wanc_b, anchor_b, anchor, 4096, 128, 256);

    // CPB bias tables -> full per-head TRANSPOSED bias maps
    cpb_mlp_k<<<(961 + 255) / 256, 256, 0, stream>>>(table_w, cw1_w, cb1_w, cw2_w, bt_w, 961);
    cpb_mlp_k<<<(1521 + 255) / 256, 256, 0, stream>>>(table_s, cw1_s1, cb1_s1, cw2_s1, bt_s1, 1521);
    cpb_mlp_k<<<(1521 + 255) / 256, 256, 0, stream>>>(table_s, cw1_s2, cb1_s2, cw2_s2, bt_s2, 1521);
    bias_expand_k<<<(4 * 65536 + 255) / 256, 256, 0, stream>>>(bt_w, index_w, bias_w, 256, 256);
    bias_expand_k<<<(4 * 65536 + 255) / 256, 256, 0, stream>>>(bt_s1, index_a2w, bias_a2w, 64, 1024);
    bias_expand_k<<<(4 * 65536 + 255) / 256, 256, 0, stream>>>(bt_s2, index_w2a, bias_w2a, 1024, 64);

    // attention
    win_attn_k<<<1024, 128, 0, stream>>>(qkv, bias_w, ls_w, yb);
    stripe_p1_k<<<256, 128, 0, stream>>>(qkv, anchor, bias_a2w, ls_s1, xm);
    stripe_p2_k<<<256, 128, 0, stream>>>(qkv, anchor, xm, bias_w2a, ls_s2, yb);

    // out = concat(xw, xs) @ proj_w + proj_b  (fp32 output)
    gemm_bf16_k<true><<<dim3(512, 4), 256, 0, stream>>>(yb, wproj_b, proj_b, d_out, 65536, 256, 256);
}
